// BBoxLSTM_24077586662081
// MI455X (gfx1250) — compile-verified, matrix operand order checked by reading, not run on hardware
//
#include <hip/hip_runtime.h>

// ---------------------------------------------------------------------------
// 2-layer LSTM (H=64) over T=32, N=8192, fused into one persistent kernel.
// Matrix work on v_wmma_f32_16x16x32_f16; recurrence state in VGPRs/LDS.
// Round 3: weight loads forced to GLOBAL addr-space (LOADcnt only, no flat),
// LICM defeated via laundered zero offset (provenance-preserving).
// ---------------------------------------------------------------------------

typedef _Float16 v16h __attribute__((ext_vector_type(16)));
typedef _Float16 v8h  __attribute__((ext_vector_type(8)));
typedef float    v8f  __attribute__((ext_vector_type(8)));

#define GAS __attribute__((address_space(1)))

#define TT     32
#define NN     8192
#define HH     64
#define GG     256
#define PITCH  72          // LDS row pitch in halfs (bank spreading), 144B
#define WAVES  2           // waves per block; each wave owns 16 rows

__device__ __forceinline__ float sig_(float x) {
    return 1.0f / (1.0f + __expf(-x));
}
__device__ __forceinline__ float tanh_(float x) {
    float e = __expf(2.0f * x);          // x>>0 -> inf -> 1 ; x<<0 -> 0 -> -1
    return 1.0f - 2.0f / (e + 1.0f);
}
__device__ __forceinline__ v8f splat8(float s) {
    v8f v;
#pragma unroll
    for (int i = 0; i < 8; ++i) v[i] = s;
    return v;
}

// Global-address-space B-tile load: guaranteed global_load_b128 (LOADcnt only).
__device__ __forceinline__ v16h loadB(const _Float16* p) {
    return *(const GAS v16h*)(const GAS _Float16*)p;
}

// Load a 16x32 f16 A-operand tile from an LDS h-buffer (row-major, PITCH).
// Lane layout per ISA: M = lane%16 ; half j -> K = kk*32 + j%8 + (j/8)*16 + (lane>=16)*8
__device__ __forceinline__ v16h loadA(const _Float16* base, int lm, int hi, int kk) {
    const _Float16* p = base + lm * PITCH + kk * 32 + hi * 8;
    v8h lo = *(const v8h*)(p);          // K = kbase .. kbase+7   (ds_load_b128)
    v8h hp = *(const v8h*)(p + 16);     // K = kbase+16 .. kbase+23
    return __builtin_shufflevector(lo, hp, 0, 1, 2, 3, 4, 5, 6, 7,
                                           8, 9, 10, 11, 12, 13, 14, 15);
}

// ---------------------------------------------------------------------------
// Prep kernel 1: pack a 256x64 f32 weight matrix into f16 B-operand tiles.
// Tile (ct,kk): 32 lanes x 16 halfs contiguous; lane holds col g=ct*16+lane%16,
// half j holds K = kk*32 + j%8 + (j/8)*16 + (lane>=16)*8.
// ---------------------------------------------------------------------------
__global__ void pack_weights(const float* __restrict__ whh0,
                             const float* __restrict__ wih1,
                             const float* __restrict__ whh1,
                             _Float16* __restrict__ outp) {
    const int bid  = blockIdx.x;         // 96 blocks: 3 matrices * 16 ct * 2 kk
    const int lane = threadIdx.x;        // 32
    const int m    = bid >> 5;
    const int rem  = bid & 31;
    const int ct   = rem >> 1;
    const int kk   = rem & 1;
    const float* W = (m == 0) ? whh0 : ((m == 1) ? wih1 : whh1);
    _Float16* o = outp + (size_t)m * 16384 + ((ct * 2 + kk) * 32 + lane) * 16;
    const int g  = ct * 16 + (lane & 15);
    const int hi = lane >> 4;
#pragma unroll
    for (int j = 0; j < 16; ++j) {
        int kloc = (j & 7) + (j >> 3) * 16 + hi * 8;
        o[j] = (_Float16)W[g * HH + kk * 32 + kloc];
    }
}

// Prep kernel 2: fused biases b = b_ih + b_hh for both layers.
__global__ void prep_bias(const float* __restrict__ bi0, const float* __restrict__ bh0,
                          const float* __restrict__ bi1, const float* __restrict__ bh1,
                          float* __restrict__ b0, float* __restrict__ b1) {
    int i = threadIdx.x;
    if (i < GG) { b0[i] = bi0[i] + bh0[i]; b1[i] = bi1[i] + bh1[i]; }
}

// ---------------------------------------------------------------------------
// Main fused LSTM kernel. One wave == one 16-row tile for all 32 timesteps.
// ---------------------------------------------------------------------------
__global__ __launch_bounds__(WAVES * 32)
void lstm_main(const float* __restrict__ bbox,
               const _Float16* __restrict__ whh0p,
               const _Float16* __restrict__ wih1p,
               const _Float16* __restrict__ whh1p,
               const float* __restrict__ b0g,
               const float* __restrict__ b1g,
               const float* __restrict__ wih0,
               const float* __restrict__ fcW,
               const float* __restrict__ fcb,
               float* __restrict__ out) {
    __shared__ __align__(16) _Float16 h0s[WAVES][16][PITCH];
    __shared__ __align__(16) _Float16 h1s[WAVES][16][PITCH];
    __shared__ float    xsm[WAVES][16][2];
    __shared__ unsigned badf[WAVES][16];
    // t-invariant per-gate-column constants, kept in LDS to relieve VGPRs
    __shared__ float cb0[GG], cb1[GG], cw0[GG], cw1[GG];

    const int tid  = threadIdx.x;
    const int w    = tid >> 5;
    const int lane = tid & 31;
    const int lm   = lane & 15;
    const int hi   = lane >> 4;
    const int rowBase = blockIdx.x * (WAVES * 16) + w * 16;

    // ---- init LDS state ----
    if (lane < 16) {
        for (int k = 0; k < PITCH; ++k) {
            h0s[w][lane][k] = (_Float16)0;
            h1s[w][lane][k] = (_Float16)0;
        }
        badf[w][lane] = 0u;
    }
    for (int i = tid; i < GG; i += WAVES * 32) {
        cb0[i] = b0g[i];
        cb1[i] = b1g[i];
        cw0[i] = wih0[i * 2 + 0];
        cw1[i] = wih0[i * 2 + 1];
    }
    __syncthreads();

    // ---- validity prepass: lane scans row lm over t in [hi*16, hi*16+16) ----
    {
        unsigned bad = 0;
        const float* p = bbox + ((size_t)(hi * 16) * NN + rowBase + lm) * 2;
        for (int t = 0; t < 16; ++t) {
            float a = p[0], b = p[1];
            if (a != a || b != b) bad = 1u;
            p += (size_t)NN * 2;
        }
        if (bad) atomicOr(&badf[w][lm], 1u);
    }
    __syncthreads();
    const unsigned myBad = badf[w][lm];    // invalid flag for row lm

    v8f c0a[4], c1a[4];
#pragma unroll
    for (int j = 0; j < 4; ++j) { c0a[j] = splat8(0.0f); c1a[j] = splat8(0.0f); }

    // ================= time recurrence =================
    for (int t = 0; t < TT; ++t) {
        // Launder a zero offset (uniform SGPR) each iteration: the weight
        // addresses become loop-variant (no LICM -> no 96KB of hoisted VGPRs
        // -> no spills) while keeping pointer provenance, so loads stay
        // global_load_b128 (LOADcnt only; no flat/DScnt coupling).
        unsigned int zoff = 0;
        asm volatile("" : "+s"(zoff));
        const _Float16* W0 = whh0p + zoff;
        const _Float16* W1 = wih1p + zoff;
        const _Float16* W2 = whh1p + zoff;

        // stage sanitized x_t for this wave's 16 rows
        if (lane < 16) {
            const float* xp = bbox + ((size_t)t * NN + rowBase + lane) * 2;
            float a = xp[0], b = xp[1];
            if (myBad) { a = 0.0f; b = 0.0f; }
            xsm[w][lane][0] = a;
            xsm[w][lane][1] = b;
        }
        __syncthreads();

        // per-row x values for accumulator init (row = r + 8*hi)
        float xr0[8], xr1[8];
#pragma unroll
        for (int r = 0; r < 8; ++r) {
            xr0[r] = xsm[w][r + 8 * hi][0];
            xr1[r] = xsm[w][r + 8 * hi][1];
        }

        // A operands: previous h0 (layer-0 recurrent input)
        v16h a0k0 = loadA(&h0s[w][0][0], lm, hi, 0);
        v16h a0k1 = loadA(&h0s[w][0][0], lm, hi, 1);
        __syncthreads();

        // ---------------- layer 0 ----------------
#pragma unroll
        for (int j = 0; j < 4; ++j) {
            v8f acc[4];
#pragma unroll
            for (int gi = 0; gi < 4; ++gi) {
                const int ct = gi * 4 + j;             // i,f,g,o gate blocks
                const float bb = cb0[ct * 16 + lm];
                const float w0 = cw0[ct * 16 + lm];
                const float w1 = cw1[ct * 16 + lm];
                v8f a;
#pragma unroll
                for (int r = 0; r < 8; ++r)
                    a[r] = bb + xr0[r] * w0 + xr1[r] * w1;
                v16h bt0 = loadB(W0 + ((ct * 2 + 0) * 32 + lane) * 16);
                v16h bt1 = loadB(W0 + ((ct * 2 + 1) * 32 + lane) * 16);
                a = __builtin_amdgcn_wmma_f32_16x16x32_f16(false, a0k0, false, bt0,
                                                           (short)0, a, false, false);
                a = __builtin_amdgcn_wmma_f32_16x16x32_f16(false, a0k1, false, bt1,
                                                           (short)0, a, false, false);
                acc[gi] = a;
            }
            v8f c = c0a[j];
            v8f hn;
#pragma unroll
            for (int r = 0; r < 8; ++r) {
                float iv = sig_(acc[0][r]);
                float fv = sig_(acc[1][r]);
                float gv = tanh_(acc[2][r]);
                float ov = sig_(acc[3][r]);
                float cn = fv * c[r] + iv * gv;
                c[r]  = cn;
                hn[r] = ov * tanh_(cn);
            }
            c0a[j] = c;
#pragma unroll
            for (int r = 0; r < 8; ++r)
                h0s[w][r + 8 * hi][j * 16 + lm] = (_Float16)hn[r];
        }
        __syncthreads();

        // A operands for layer 1: fresh h0 and previous h1 (loaded late to
        // keep their live range out of layer 0)
        v16h ank0 = loadA(&h0s[w][0][0], lm, hi, 0);
        v16h ank1 = loadA(&h0s[w][0][0], lm, hi, 1);
        v16h a1k0 = loadA(&h1s[w][0][0], lm, hi, 0);
        v16h a1k1 = loadA(&h1s[w][0][0], lm, hi, 1);

        // ---------------- layer 1 ----------------
#pragma unroll
        for (int j = 0; j < 4; ++j) {
            v8f acc[4];
#pragma unroll
            for (int gi = 0; gi < 4; ++gi) {
                const int ct = gi * 4 + j;
                v8f a = splat8(cb1[ct * 16 + lm]);
                v16h bi0 = loadB(W1 + ((ct * 2 + 0) * 32 + lane) * 16);
                v16h bi1 = loadB(W1 + ((ct * 2 + 1) * 32 + lane) * 16);
                v16h bh0 = loadB(W2 + ((ct * 2 + 0) * 32 + lane) * 16);
                v16h bh1 = loadB(W2 + ((ct * 2 + 1) * 32 + lane) * 16);
                a = __builtin_amdgcn_wmma_f32_16x16x32_f16(false, ank0, false, bi0,
                                                           (short)0, a, false, false);
                a = __builtin_amdgcn_wmma_f32_16x16x32_f16(false, ank1, false, bi1,
                                                           (short)0, a, false, false);
                a = __builtin_amdgcn_wmma_f32_16x16x32_f16(false, a1k0, false, bh0,
                                                           (short)0, a, false, false);
                a = __builtin_amdgcn_wmma_f32_16x16x32_f16(false, a1k1, false, bh1,
                                                           (short)0, a, false, false);
                acc[gi] = a;
            }
            v8f c = c1a[j];
            v8f hn;
#pragma unroll
            for (int r = 0; r < 8; ++r) {
                float iv = sig_(acc[0][r]);
                float fv = sig_(acc[1][r]);
                float gv = tanh_(acc[2][r]);
                float ov = sig_(acc[3][r]);
                float cn = fv * c[r] + iv * gv;
                c[r]  = cn;
                hn[r] = ov * tanh_(cn);
            }
            c1a[j] = c;
#pragma unroll
            for (int r = 0; r < 8; ++r)
                h1s[w][r + 8 * hi][j * 16 + lm] = (_Float16)hn[r];
        }
        __syncthreads();
    }

    // ---- FC head: pred = h1_final @ fc_W^T + fc_b ; NaN for invalid rows ----
    if (lane < 16) {
        const int row = rowBase + lane;
        float s0 = fcb[0], s1 = fcb[1];
#pragma unroll
        for (int k = 0; k < HH; ++k) {
            float hv = (float)h1s[w][lane][k];
            s0 += hv * fcW[k];
            s1 += hv * fcW[HH + k];
        }
        if (myBad) { s0 = __builtin_nanf(""); s1 = __builtin_nanf(""); }
        out[row * 2 + 0] = s0;
        out[row * 2 + 1] = s1;
    }
}

// ---------------------------------------------------------------------------
extern "C" void kernel_launch(void* const* d_in, const int* in_sizes, int n_in,
                              void* d_out, int out_size, void* d_ws, size_t ws_size,
                              hipStream_t stream) {
    const float* bbox = (const float*)d_in[0];
    const float* Wih0 = (const float*)d_in[1];
    const float* Whh0 = (const float*)d_in[2];
    const float* bih0 = (const float*)d_in[3];
    const float* bhh0 = (const float*)d_in[4];
    const float* Wih1 = (const float*)d_in[5];
    const float* Whh1 = (const float*)d_in[6];
    const float* bih1 = (const float*)d_in[7];
    const float* bhh1 = (const float*)d_in[8];
    const float* fcW  = (const float*)d_in[9];
    const float* fcb  = (const float*)d_in[10];
    float* out = (float*)d_out;

    // workspace: 3 packed f16 matrices (3*32KB) + fused biases (2*1KB)
    _Float16* pk = (_Float16*)d_ws;
    float* b0 = (float*)((char*)d_ws + (size_t)3 * 16384 * sizeof(_Float16));
    float* b1 = b0 + GG;

    pack_weights<<<96, 32, 0, stream>>>(Whh0, Wih1, Whh1, pk);
    prep_bias<<<1, 256, 0, stream>>>(bih0, bhh0, bih1, bhh1, b0, b1);
    lstm_main<<<NN / (WAVES * 16), WAVES * 32, 0, stream>>>(
        bbox, pk, pk + 16384, pk + 32768, b0, b1, Wih0, fcW, fcb, out);
}
